// CenterHeadGroupSbnet_39522289058097
// MI455X (gfx1250) — matrix operand order repaired and checked by reading, not compiled, not run on hardware
//
#include <hip/hip_runtime.h>
#include <hip/hip_bf16.h>

// ---------------- problem constants ----------------
#define BATCH   2
#define HH      180
#define WW      180
#define TC      18      // SBNet tiles per dim (10x10 px tiles)
#define C_IN    512
#define SC      64      // shared conv channels
#define HM1C    384
#define AT1C    1920
#define HM_G    6
#define HM_CPG  2       // 12 heatmap channels
#define AT_G    30
#define AT_CPG  3       // 90 attr channels
#define OUTC    102     // 12 + 90
#define XT      12      // 16-px x-strips (grouped kernel)
#define XT2     6       // 32-px x-strips (dense conv kernel)
#define EPSBN   1e-5f

// ---------------- CDNA5 async LDS path (guarded) ----------------
#if defined(__has_builtin)
#if __has_builtin(__builtin_amdgcn_global_load_async_to_lds_b128)
#define HAVE_ASYNC_LDS 1
#endif
#endif

typedef int v4i_gcc __attribute__((vector_size(16)));
typedef __attribute__((address_space(1))) void*    as1void_p;
typedef __attribute__((address_space(3))) void*    as3void_p;
typedef __attribute__((address_space(1))) v4i_gcc* as1v4i_p;
typedef __attribute__((address_space(3))) v4i_gcc* as3v4i_p;

__device__ __forceinline__ void wait_asynccnt0() {
#if defined(__has_builtin) && __has_builtin(__builtin_amdgcn_s_wait_asynccnt)
    __builtin_amdgcn_s_wait_asynccnt(0);
#else
    asm volatile("s_wait_asynccnt 0" ::: "memory");
#endif
}

// ---------------- WMMA fragment types ----------------
typedef __attribute__((ext_vector_type(16))) __bf16 v16bf;
typedef __attribute__((ext_vector_type(8)))  __bf16 v8bf;
typedef __attribute__((ext_vector_type(8)))  float  v8f;

__device__ __forceinline__ unsigned short f2bf(float f) {
    union { float f; unsigned u; } x; x.f = f;
    unsigned r = (x.u + 0x7FFFu + ((x.u >> 16) & 1u)) >> 16;
    return (unsigned short)r;
}

__device__ __forceinline__ v16bf cat16(v8bf lo, v8bf hi) {
    v16bf r;
#pragma unroll
    for (int i = 0; i < 8; ++i) { r[i] = lo[i]; r[i + 8] = hi[i]; }
    return r;
}

// ---------------- workspace layout (bytes, 256-aligned) ----------------
static constexpr size_t O_XG  = 0;                       // 66,355,200
static constexpr size_t O_SH  = O_XG  + 66355200ull;     //  8,294,400
static constexpr size_t O_HB  = O_SH  + 8294400ull;      // 49,766,400
static constexpr size_t O_AB  = O_HB  + 49766400ull;     // 248,832,000
static constexpr size_t O_WPS = O_AB  + 248832000ull;    // 589,824
static constexpr size_t O_WPH = O_WPS + 589824ull;       // 442,368
static constexpr size_t O_WPA = O_WPH + 442368ull;       // 2,211,840
static constexpr size_t O_WGH = O_WPA + 2211840ull;      // 110,592
static constexpr size_t O_WGA = O_WGH + 110592ull;       // 552,960
static constexpr size_t O_BNS = O_WGA + 552960ull;       // 128 f32
static constexpr size_t O_BNH = O_BNS + 512ull;          // 768 f32
static constexpr size_t O_BNA = O_BNH + 3072ull;         // 3840 f32

// ---------------- input prep: NCHW f32 -> NHWC bf16, mask folded ----------------
__global__ __launch_bounds__(256) void k_prep_input(
    const float* __restrict__ x, const int* __restrict__ tmask,
    unsigned short* __restrict__ xg)
{
    long long i = (long long)blockIdx.x * 256 + threadIdx.x;
    const long long total = (long long)BATCH * HH * WW * C_IN;
    if (i >= total) return;
    int c  = (int)(i & (C_IN - 1));
    long long r = i >> 9;
    int xp = (int)(r % WW); r /= WW;
    int y  = (int)(r % HH); r /= HH;
    int b  = (int)r;
    int act = tmask[(b * TC + y / 10) * TC + xp / 10];
    float v = act ? x[(((long long)b * C_IN + c) * HH + y) * WW + xp] : 0.f;
    xg[i] = f2bf(v);
}

// ---------------- weight prepack (dense): OIHW f32 -> [tap][kc][oc][32] bf16 ----------------
__global__ __launch_bounds__(256) void k_pack_w(
    const float* __restrict__ w, unsigned short* __restrict__ wp,
    int Cin, int Cout)
{
    int KC = Cin >> 5;
    long long total = 9ll * KC * Cout * 32;
    long long i = (long long)blockIdx.x * 256 + threadIdx.x;
    if (i >= total) return;
    int k  = (int)(i & 31);
    long long r = i >> 5;
    int oc = (int)(r % Cout); r /= Cout;
    int kc = (int)(r % KC);
    int t  = (int)(r / KC);
    int ic = kc * 32 + k;
    wp[i] = f2bf(w[((long long)oc * Cin + ic) * 9 + t]);
}

// ---------------- weight prepack (grouped): -> [g][tap][kc=2][16][32] bf16 ----------------
__global__ __launch_bounds__(256) void k_pack_wg(
    const float* __restrict__ w, unsigned short* __restrict__ wp,
    int cpg, int groups)
{
    long long total = (long long)groups * 9 * 2 * 16 * 32;
    long long i = (long long)blockIdx.x * 256 + threadIdx.x;
    if (i >= total) return;
    int k  = (int)(i & 31);
    long long r = i >> 5;
    int n  = (int)(r & 15); r >>= 4;
    int kc = (int)(r & 1);  r >>= 1;
    int t  = (int)(r % 9);
    int g  = (int)(r / 9);
    float v = 0.f;
    if (n < cpg) {
        int oc = g * cpg + n;
        int ic = kc * 32 + k;
        v = w[((long long)oc * 64 + ic) * 9 + t];
    }
    wp[i] = f2bf(v);
}

// ---------------- BN fold ----------------
__global__ __launch_bounds__(256) void k_bnfold(
    const float* __restrict__ g, const float* __restrict__ b,
    const float* __restrict__ m, const float* __restrict__ v,
    float* __restrict__ scale, float* __restrict__ shift, int n)
{
    int i = blockIdx.x * 256 + threadIdx.x;
    if (i >= n) return;
    float s = g[i] * rsqrtf(v[i] + EPSBN);
    scale[i] = s;
    shift[i] = b[i] - m[i] * s;
}

// ---------------- dense conv3x3 + BN + ReLU + mask, implicit GEMM via WMMA ----------------
// block: 128 threads (4 waves). tile: 32-px x-strip x 64 out channels.
// each wave: 16 oc x 32 px -> 2 accumulators, B fragment reused across both.
// gridDim.x = B*H*XT2, gridDim.y = CoutTot/64.
__global__ __launch_bounds__(128) void k_conv_wmma(
    const unsigned short* __restrict__ xin,  // NHWC bf16 [B][H][W][Cin]
    const unsigned short* __restrict__ wp,   // [9][KC][CoutTot][32] bf16
    const float* __restrict__ scale, const float* __restrict__ shift,
    const int* __restrict__ tmask,
    unsigned short* __restrict__ yout,       // NHWC bf16 [B][H][W][CoutTot]
    int Cin, int CoutTot)
{
    int bx = blockIdx.x;
    int xt = bx % XT2;  int y = (bx / XT2) % HH;  int b = bx / (XT2 * HH);
    int x0 = xt * 32;
    int octile = blockIdx.y * 64;
    int tid = threadIdx.x, wave = tid >> 5, lane = tid & 31;
    int KC = Cin >> 5;

    const int* tmrow = tmask + (b * TC + y / 10) * TC;
    int te = x0 + 31; if (te > WW - 1) te = WW - 1;
    int anyActive = 0;
    for (int tt = x0 / 10; tt <= te / 10; ++tt) anyActive |= tmrow[tt];

    if (!anyActive) {
        // SBNet skip: strip fully inactive -> zero-fill our oc tile (32px x 64oc).
        for (int i = tid; i < 32 * 64; i += 128) {
            int mm = i >> 6, cc = i & 63, xx = x0 + mm;
            if (xx < WW)
                yout[(((size_t)b * HH + y) * WW + xx) * CoutTot + octile + cc] = 0;
        }
        return;
    }

    // A tile: 32 px x 32 k bf16, 80B row stride (keeps 16B chunk alignment)
    __shared__ __align__(16) unsigned short Asm[32][40];

    int ln = lane & 15, hi = lane >> 4;
    int oc0 = octile + wave * 16;
    v8f acc0 = {};   // pixels x0+0..15
    v8f acc1 = {};   // pixels x0+16..31

    for (int t = 0; t < 9; ++t) {
        int dy = t / 3 - 1, dx = t % 3 - 1;
        int iy = y + dy;
        bool rowOK = (iy >= 0) && (iy < HH);
        for (int kc = 0; kc < KC; ++kc) {
            __syncthreads();   // previous-iteration reads done before overwrite
            {   // stage A: 128 threads x 16B (one async b128 each)
                int mm = tid >> 2;            // 0..31 pixel row
                int kk = (tid & 3) * 8;       // 0,8,16,24 (bf16 units)
                int ix = x0 + mm + dx;
                if (rowOK && ix >= 0 && ix < WW) {
                    const unsigned short* src =
                        xin + (((size_t)b * HH + iy) * WW + ix) * Cin + kc * 32 + kk;
#ifdef HAVE_ASYNC_LDS
                    __builtin_amdgcn_global_load_async_to_lds_b128(
                        (as1v4i_p)(as1void_p)(void*)(const void*)src,
                        (as3v4i_p)(as3void_p)(void*)&Asm[mm][kk],
                        0, 0);
#else
                    *(ulonglong2*)&Asm[mm][kk] = *(const ulonglong2*)src;
#endif
                } else {
                    ulonglong2 z; z.x = 0; z.y = 0;
                    *(ulonglong2*)&Asm[mm][kk] = z;
                }
            }
#ifdef HAVE_ASYNC_LDS
            wait_asynccnt0();
#endif
            __syncthreads();

            // B fragment: two 16B reads from prepacked global (L2 resident), reused 2x
            const unsigned short* bp =
                wp + ((((size_t)t * KC + kc) * CoutTot) + (oc0 + ln)) * 32 + (hi ? 8 : 0);
            __builtin_prefetch(bp + CoutTot * 32, 0, 1);
            v8bf blo = *(const v8bf*)bp;
            v8bf bhi = *(const v8bf*)(bp + 16);
            v16bf bfr = cat16(blo, bhi);

            // A fragments (ISA 16-bit A 16x32 lane layout): two 16B LDS reads each
            const char* ar0 = (const char*)&Asm[ln][0];
            v16bf af0 = cat16(*(const v8bf*)(ar0 + (hi ? 16 : 0)),
                              *(const v8bf*)(ar0 + (hi ? 48 : 32)));
            acc0 = __builtin_amdgcn_wmma_f32_16x16x32_bf16(
                false, af0, false, bfr, (short)0, acc0, false, false);

            const char* ar1 = (const char*)&Asm[16 + ln][0];
            v16bf af1 = cat16(*(const v8bf*)(ar1 + (hi ? 16 : 0)),
                              *(const v8bf*)(ar1 + (hi ? 48 : 32)));
            acc1 = __builtin_amdgcn_wmma_f32_16x16x32_bf16(
                false, af1, false, bfr, (short)0, acc1, false, false);
        }
    }

    // fused BN + ReLU + tile mask, store NHWC bf16 (C/D layout: M=r+8*hi, N=ln)
    int c = oc0 + ln;
    float sc = scale[c], sf = shift[c];
#pragma unroll
    for (int r = 0; r < 8; ++r) {
        int mm = r + (hi << 3);
        {   // acc0 -> pixels x0 + mm
            int xx = x0 + mm;
            if (xx < WW) {
                int act = tmrow[xx / 10];
                float v = acc0[r] * sc + sf;
                v = (act && v > 0.f) ? v : 0.f;
                yout[(((size_t)b * HH + y) * WW + xx) * CoutTot + c] = f2bf(v);
            }
        }
        {   // acc1 -> pixels x0 + 16 + mm
            int xx = x0 + 16 + mm;
            if (xx < WW) {
                int act = tmrow[xx / 10];
                float v = acc1[r] * sc + sf;
                v = (act && v > 0.f) ? v : 0.f;
                yout[(((size_t)b * HH + y) * WW + xx) * CoutTot + c] = f2bf(v);
            }
        }
    }
}

// ---------------- grouped conv3x3 + bias + mask -> final f32 NCHW output ----------------
// block: 32 threads (1 wave). gridDim.x = B*H*XT, gridDim.y = groups.
__global__ __launch_bounds__(32) void k_convg_wmma(
    const unsigned short* __restrict__ xin,  // NHWC bf16 [B][H][W][CinTot]
    const unsigned short* __restrict__ wpg,  // [G][9][2][16][32] bf16
    const float* __restrict__ bias, const int* __restrict__ tmask,
    float* __restrict__ out,                 // NCHW f32 [B][OUTC][H][W]
    int CinTot, int cpg, int cbase)
{
    int bx = blockIdx.x;
    int xt = bx % XT;  int y = (bx / XT) % HH;  int b = bx / (XT * HH);
    int x0 = xt * 16;
    int g  = blockIdx.y;
    int lane = threadIdx.x, ln = lane & 15, hi = lane >> 4;

    const int* tmrow = tmask + (b * TC + y / 10) * TC;
    int xhi = x0 + 15; if (xhi > WW - 1) xhi = WW - 1;
    bool anyActive = (tmrow[x0 / 10] | tmrow[xhi / 10]) != 0;

    if (!anyActive) {
        if (ln < cpg) {
            int c = cbase + g * cpg + ln;
#pragma unroll
            for (int r = 0; r < 8; ++r) {
                int mm = r + (hi << 3), xx = x0 + mm;
                if (xx < WW)
                    out[(((size_t)b * OUTC + c) * HH + y) * WW + xx] = 0.f;
            }
        }
        return;
    }

    size_t ic0 = (size_t)g * 64;
    v8f acc = {};

    for (int t = 0; t < 9; ++t) {
        int dy = t / 3 - 1, dx = t % 3 - 1;
        int iy = y + dy;
        bool rowOK = (iy >= 0) && (iy < HH);
#pragma unroll
        for (int kc = 0; kc < 2; ++kc) {
            int ix = x0 + ln + dx;
            v8bf alo = {}, ahi = {};
            if (rowOK && ix >= 0 && ix < WW) {
                const unsigned short* src =
                    xin + (((size_t)b * HH + iy) * WW + ix) * CinTot + ic0 + kc * 32;
                alo = *(const v8bf*)(src + (hi ? 8 : 0));
                ahi = *(const v8bf*)(src + (hi ? 24 : 16));
            }
            v16bf afr = cat16(alo, ahi);

            const unsigned short* bp =
                wpg + (((((size_t)g * 9 + t) * 2 + kc) * 16 + ln) * 32) + (hi ? 8 : 0);
            v8bf blo = *(const v8bf*)bp;
            v8bf bhi = *(const v8bf*)(bp + 16);
            v16bf bfr = cat16(blo, bhi);

            acc = __builtin_amdgcn_wmma_f32_16x16x32_bf16(
                false, afr, false, bfr, (short)0, acc, false, false);
        }
    }

    if (ln < cpg) {
        int c = cbase + g * cpg + ln;
        float bs = bias[g * cpg + ln];
#pragma unroll
        for (int r = 0; r < 8; ++r) {
            int mm = r + (hi << 3), xx = x0 + mm;
            if (xx < WW) {
                int act = tmrow[xx / 10];
                out[(((size_t)b * OUTC + c) * HH + y) * WW + xx] =
                    act ? (acc[r] + bs) : 0.f;
            }
        }
    }
}

// ---------------- host launcher ----------------
extern "C" void kernel_launch(void* const* d_in, const int* in_sizes, int n_in,
                              void* d_out, int out_size, void* d_ws, size_t ws_size,
                              hipStream_t stream) {
    const float* x        = (const float*)d_in[0];
    const int*   tmask    = (const int*)  d_in[1];
    const float* w_shared = (const float*)d_in[2];
    const float* g_s = (const float*)d_in[3], *b_s = (const float*)d_in[4];
    const float* m_s = (const float*)d_in[5], *v_s = (const float*)d_in[6];
    const float* w_hm1 = (const float*)d_in[7];
    const float* g_h = (const float*)d_in[8],  *b_h = (const float*)d_in[9];
    const float* m_h = (const float*)d_in[10], *v_h = (const float*)d_in[11];
    const float* w_hm2    = (const float*)d_in[12];
    const float* bias_hm  = (const float*)d_in[13];
    const float* w_attr1  = (const float*)d_in[14];
    const float* g_a = (const float*)d_in[15], *b_a = (const float*)d_in[16];
    const float* m_a = (const float*)d_in[17], *v_a = (const float*)d_in[18];
    const float* w_attr2   = (const float*)d_in[19];
    const float* bias_attr = (const float*)d_in[20];
    float* out = (float*)d_out;

    char* ws = (char*)d_ws;
    unsigned short* xg  = (unsigned short*)(ws + O_XG);
    unsigned short* shb = (unsigned short*)(ws + O_SH);
    unsigned short* hb  = (unsigned short*)(ws + O_HB);
    unsigned short* ab  = (unsigned short*)(ws + O_AB);
    unsigned short* wps = (unsigned short*)(ws + O_WPS);
    unsigned short* wph = (unsigned short*)(ws + O_WPH);
    unsigned short* wpa = (unsigned short*)(ws + O_WPA);
    unsigned short* wgh = (unsigned short*)(ws + O_WGH);
    unsigned short* wga = (unsigned short*)(ws + O_WGA);
    float* bns = (float*)(ws + O_BNS);
    float* bnh = (float*)(ws + O_BNH);
    float* bna = (float*)(ws + O_BNA);

    {
        long long total = (long long)BATCH * HH * WW * C_IN;
        k_prep_input<<<(unsigned)((total + 255) / 256), 256, 0, stream>>>(x, tmask, xg);
    }
    k_pack_w<<<(9 * 16 * 64 * 32 + 255) / 256, 256, 0, stream>>>(w_shared, wps, C_IN, SC);
    k_pack_w<<<(9 * 2 * HM1C * 32 + 255) / 256, 256, 0, stream>>>(w_hm1, wph, SC, HM1C);
    k_pack_w<<<(9 * 2 * AT1C * 32 + 255) / 256, 256, 0, stream>>>(w_attr1, wpa, SC, AT1C);
    k_pack_wg<<<(HM_G * 9 * 2 * 16 * 32 + 255) / 256, 256, 0, stream>>>(w_hm2, wgh, HM_CPG, HM_G);
    k_pack_wg<<<(AT_G * 9 * 2 * 16 * 32 + 255) / 256, 256, 0, stream>>>(w_attr2, wga, AT_CPG, AT_G);
    k_bnfold<<<1, 256, 0, stream>>>(g_s, b_s, m_s, v_s, bns, bns + SC, SC);
    k_bnfold<<<(HM1C + 255) / 256, 256, 0, stream>>>(g_h, b_h, m_h, v_h, bnh, bnh + HM1C, HM1C);
    k_bnfold<<<(AT1C + 255) / 256, 256, 0, stream>>>(g_a, b_a, m_a, v_a, bna, bna + AT1C, AT1C);

    dim3 blk(128);
    unsigned gx2 = BATCH * HH * XT2;  // 2160 (32-px strips)
    unsigned gx  = BATCH * HH * XT;   // 4320 (16-px strips)
    k_conv_wmma<<<dim3(gx2, 1), blk, 0, stream>>>(xg, wps, bns, bns + SC, tmask, shb, C_IN, SC);
    k_conv_wmma<<<dim3(gx2, HM1C / 64), blk, 0, stream>>>(shb, wph, bnh, bnh + HM1C, tmask, hb, SC, HM1C);
    k_conv_wmma<<<dim3(gx2, AT1C / 64), blk, 0, stream>>>(shb, wpa, bna, bna + AT1C, tmask, ab, SC, AT1C);
    k_convg_wmma<<<dim3(gx, HM_G), dim3(32), 0, stream>>>(hb, wgh, bias_hm, tmask, out, HM1C, HM_CPG, 0);
    k_convg_wmma<<<dim3(gx, AT_G), dim3(32), 0, stream>>>(ab, wga, bias_attr, tmask, out, AT1C, AT_CPG, 12);
}
